// EventTempRel_HGRU_static_14886356648166
// MI455X (gfx1250) — matrix operand order repaired and checked by reading, not compiled, not run on hardware
//
#include <hip/hip_runtime.h>
#include <hip/hip_bf16.h>
#include <math.h>

typedef __attribute__((ext_vector_type(16))) _Float16 v16h;
typedef __attribute__((ext_vector_type(8)))  _Float16 v8h;
typedef __attribute__((ext_vector_type(8)))  float    v8f;

#define B_    256
#define T_    128
#define DIN   768
#define H_    128
#define DOUT_ 64
#define NCLS  4
#define CSD   64
#define N3H   384

__device__ __forceinline__ float wred(float v) {
#pragma unroll
  for (int off = 16; off > 0; off >>= 1) v += __shfl_xor(v, off, 32);
  return v;
}
__device__ __forceinline__ float clmin(float x) { return fmaxf(x, 1e-15f); }
__device__ __forceinline__ float artanh_c(float x) {
  x = fminf(fmaxf(x, -1.0f + 1e-5f), 1.0f - 1e-5f);
  return atanhf(x);
}

// ---------------- Kernel 1: expmap0 + f16 cast + norm (xn = tanh(||u||)) ----
__global__ __launch_bounds__(256) void k_prep(const float* __restrict__ seq,
                                              _Float16* __restrict__ projh,
                                              float* __restrict__ xn) {
  const int row  = blockIdx.x * 8 + (threadIdx.x >> 5);
  const int lane = threadIdx.x & 31;
  const float* src = seq + (size_t)row * DIN;
  float v[24]; float s = 0.f;
#pragma unroll
  for (int j = 0; j < 24; ++j) { v[j] = src[lane + 32 * j]; s += v[j] * v[j]; }
  s = wred(s);
  float un = clmin(sqrtf(s));
  float sc = tanhf(un) / un;
  _Float16* dst = projh + (size_t)row * DIN;
#pragma unroll
  for (int j = 0; j < 24; ++j) dst[lane + 32 * j] = (_Float16)(v[j] * sc);
  if (lane == 0) xn[row] = clmin(tanhf(un));
}

// ---------------- Kernel 2: pack weights to f16, concatenated z|r|h ---------
__global__ __launch_bounds__(256) void k_pack(const float* __restrict__ wz, const float* __restrict__ wr,
                                              const float* __restrict__ wh, const float* __restrict__ uz,
                                              const float* __restrict__ ur, const float* __restrict__ uh,
                                              _Float16* __restrict__ Ucat, _Float16* __restrict__ Wcat) {
  const int idx = blockIdx.x * 256 + threadIdx.x;
  const int NU = N3H * DIN;
  if (idx < NU) {
    int r = idx / DIN, k = idx % DIN;
    const float* s = (r < H_) ? uz : (r < 2 * H_) ? ur : uh;
    Ucat[idx] = (_Float16)s[(r % H_) * DIN + k];
  } else {
    int i2 = idx - NU;
    if (i2 < N3H * H_) {
      int r = i2 / H_, k = i2 % H_;
      const float* s = (r < H_) ? wz : (r < 2 * H_) ? wr : wh;
      Wcat[i2] = (_Float16)s[(r % H_) * H_ + k];
    }
  }
}

// ---------------- Kernel 3: big WMMA GEMM  Ux = proj @ Ucat^T ---------------
// One block per 16-row M-tile; A (16x768 f16 = 24 KB) staged once in LDS and
// shared by 8 waves x 3 N-tiles each (covers all 384 output columns).
__global__ __launch_bounds__(256) void k_gemm_ux(const _Float16* __restrict__ Ah,
                                                 const _Float16* __restrict__ Bh,
                                                 float* __restrict__ Cout) {
  __shared__ _Float16 sA[16 * DIN];  // 24 KB, row-major [16][768]
  const int tid = threadIdx.x, wave = tid >> 5, lane = tid & 31;
  const int l16 = lane & 15, g16 = lane >> 4;
  const int mt = blockIdx.x;

  // Stage A: 16 consecutive rows are one contiguous 24 KB block.
  {
    const _Float16* src = Ah + (size_t)mt * 16 * DIN;
#pragma unroll
    for (int c = 0; c < 6; ++c) {
      const int idx8 = (c * 256 + tid) * 8;  // in halves (16B chunks)
      *(v8h*)(sA + idx8) = *(const v8h*)(src + idx8);
    }
  }
  __syncthreads();

  const int nt0 = wave * 3;
  const _Float16* brow0 = Bh + (size_t)((nt0 + 0) * 16 + l16) * DIN;
  const _Float16* brow1 = Bh + (size_t)((nt0 + 1) * 16 + l16) * DIN;
  const _Float16* brow2 = Bh + (size_t)((nt0 + 2) * 16 + l16) * DIN;
  v8f acc0 = {}, acc1 = {}, acc2 = {};

#pragma unroll 4
  for (int ks = 0; ks < 24; ++ks) {
    const _Float16* ap = sA + l16 * DIN + ks * 32 + 16 * g16;
    v8h a0 = *(const v8h*)ap, a1 = *(const v8h*)(ap + 8);
    v16h A;
#pragma unroll
    for (int i = 0; i < 8; ++i) { A[i] = a0[i]; A[8 + i] = a1[i]; }
    if (ks + 1 < 24) {
      __builtin_prefetch(brow0 + (ks + 1) * 32, 0, 1);
      __builtin_prefetch(brow1 + (ks + 1) * 32, 0, 1);
      __builtin_prefetch(brow2 + (ks + 1) * 32, 0, 1);
    }
    {
      const _Float16* bp = brow0 + ks * 32 + 16 * g16;
      v8h b0 = *(const v8h*)bp, b1 = *(const v8h*)(bp + 8);
      v16h Bm;
#pragma unroll
      for (int i = 0; i < 8; ++i) { Bm[i] = b0[i]; Bm[8 + i] = b1[i]; }
      acc0 = __builtin_amdgcn_wmma_f32_16x16x32_f16(false, A, false, Bm, (short)0, acc0, false, false);
    }
    {
      const _Float16* bp = brow1 + ks * 32 + 16 * g16;
      v8h b0 = *(const v8h*)bp, b1 = *(const v8h*)(bp + 8);
      v16h Bm;
#pragma unroll
      for (int i = 0; i < 8; ++i) { Bm[i] = b0[i]; Bm[8 + i] = b1[i]; }
      acc1 = __builtin_amdgcn_wmma_f32_16x16x32_f16(false, A, false, Bm, (short)0, acc1, false, false);
    }
    {
      const _Float16* bp = brow2 + ks * 32 + 16 * g16;
      v8h b0 = *(const v8h*)bp, b1 = *(const v8h*)(bp + 8);
      v16h Bm;
#pragma unroll
      for (int i = 0; i < 8; ++i) { Bm[i] = b0[i]; Bm[8 + i] = b1[i]; }
      acc2 = __builtin_amdgcn_wmma_f32_16x16x32_f16(false, A, false, Bm, (short)0, acc2, false, false);
    }
  }
#pragma unroll
  for (int i = 0; i < 8; ++i) {
    const size_t rowo = (size_t)(mt * 16 + i + 8 * g16) * N3H;
    Cout[rowo + (nt0 + 0) * 16 + l16] = acc0[i];
    Cout[rowo + (nt0 + 1) * 16 + l16] = acc1[i];
    Cout[rowo + (nt0 + 2) * 16 + l16] = acc2[i];
  }
}

// ---------------- Kernel 4: per-chunk mobius_matvec rescale -----------------
__global__ __launch_bounds__(256) void k_scale_ux(float* __restrict__ Ux,
                                                  const float* __restrict__ xn) {
  const int widx = blockIdx.x * 8 + (threadIdx.x >> 5);
  const int lane = threadIdx.x & 31;
  const int row = widx / 3, c = widx % 3;
  float* p = Ux + (size_t)row * N3H + c * H_;
  float v[4]; float s = 0.f;
#pragma unroll
  for (int j = 0; j < 4; ++j) { v[j] = p[lane + 32 * j]; s += v[j] * v[j]; }
  s = wred(s);
  float mxn = clmin(sqrtf(s));
  float x = xn[row];
  float sc = tanhf(mxn / x * artanh_c(x)) / mxn;
#pragma unroll
  for (int j = 0; j < 4; ++j) p[lane + 32 * j] = v[j] * sc;
}

// ---------------- Kernel 5: persistent hyperbolic GRU scan ------------------
// 16 batch rows / workgroup (256 threads = 8 waves, wave32), h in LDS.
// All recurrent-weight WMMA B-fragments are hoisted into registers (12 v16h
// = 96 VGPRs/lane) so the steady-state step does no global weight traffic.
__global__ __launch_bounds__(256) void k_scan(const float* __restrict__ Ux,
                                              const _Float16* __restrict__ Wcat,
                                              const float* __restrict__ bz, const float* __restrict__ brv,
                                              const float* __restrict__ bh,
                                              const float* __restrict__ mask1, const float* __restrict__ mask2,
                                              float* __restrict__ ubuf, float* __restrict__ vbuf) {
  __shared__ float    sh_h[16][H_];
  __shared__ _Float16 sh_a[16][H_];
  __shared__ float    sh_g[16][2 * H_];
  __shared__ float    sh_b[3][H_];
  __shared__ float    sh_bn2[3];

  const int tid = threadIdx.x, wave = tid >> 5, lane = tid & 31;
  const int l16 = lane & 15, g16 = lane >> 4;

  for (int i = tid; i < H_; i += 256) { sh_b[0][i] = bz[i]; sh_b[1][i] = brv[i]; sh_b[2][i] = bh[i]; }
  for (int i = tid; i < 16 * H_; i += 256) { (&sh_h[0][0])[i] = 0.f; (&sh_a[0][0])[i] = (_Float16)0.f; }
  __syncthreads();
  if (wave < 3) {
    float s = 0.f;
#pragma unroll
    for (int j = 0; j < 4; ++j) { float b = sh_b[wave][lane + 32 * j]; s += b * b; }
    s = wred(s);
    if (lane == 0) sh_bn2[wave] = s;
  }
  __syncthreads();

  // Hoist W fragments: GEMM1 needs tiles nt = wave*2 + q (rows 0..255 of Wcat),
  // GEMM2 needs tile nt = wave (rows 256..383).
  v16h Bw1[2][4];
  v16h Bw2[4];
#pragma unroll
  for (int q = 0; q < 2; ++q) {
#pragma unroll
    for (int ks = 0; ks < 4; ++ks) {
      const _Float16* bp = Wcat + (size_t)((wave * 2 + q) * 16 + l16) * H_ + ks * 32 + 16 * g16;
      v8h b0 = *(const v8h*)bp, b1 = *(const v8h*)(bp + 8);
#pragma unroll
      for (int i = 0; i < 8; ++i) { Bw1[q][ks][i] = b0[i]; Bw1[q][ks][8 + i] = b1[i]; }
    }
  }
#pragma unroll
  for (int ks = 0; ks < 4; ++ks) {
    const _Float16* bp = Wcat + (size_t)(2 * H_ + wave * 16 + l16) * H_ + ks * 32 + 16 * g16;
    v8h b0 = *(const v8h*)bp, b1 = *(const v8h*)(bp + 8);
#pragma unroll
    for (int i = 0; i < 8; ++i) { Bw2[ks][i] = b0[i]; Bw2[ks][8 + i] = b1[i]; }
  }

  const int r0 = wave * 2;
  float z_reg[2][4];
  float rhn_reg[2], hn2_reg[2];
  float u_acc[2][4] = {}, v_acc[2][4] = {};

  for (int t = 0; t < T_; ++t) {
    // ---- GEMM1: sh_g[:,0:256] = h_f16 @ [Wz;Wr]^T  (2 tiles per wave) ----
#pragma unroll
    for (int q = 0; q < 2; ++q) {
      const int nt = wave * 2 + q;
      v8f acc = {};
#pragma unroll
      for (int ks = 0; ks < 4; ++ks) {
        const _Float16* ap = &sh_a[l16][ks * 32 + 16 * g16];
        v8h a0 = *(const v8h*)ap, a1 = *(const v8h*)(ap + 8);
        v16h A;
#pragma unroll
        for (int i = 0; i < 8; ++i) { A[i] = a0[i]; A[8 + i] = a1[i]; }
        acc = __builtin_amdgcn_wmma_f32_16x16x32_f16(false, A, false, Bw1[q][ks], (short)0, acc, false, false);
      }
#pragma unroll
      for (int i = 0; i < 8; ++i) sh_g[i + 8 * g16][nt * 16 + l16] = acc[i];
    }
    __syncthreads();

    // ---- E1: gates z,r + rh (Mobius algebra, wave reductions) ----
#pragma unroll
    for (int rr = 0; rr < 2; ++rr) {
      const int r = r0 + rr, bi = blockIdx.x * 16 + r;
      const float* uxrow = Ux + (size_t)(bi * T_ + t) * N3H;
      float h4[4]; float hn2 = 0.f;
#pragma unroll
      for (int j = 0; j < 4; ++j) { h4[j] = sh_h[r][lane + 32 * j]; hn2 += h4[j] * h4[j]; }
      hn2 = wred(hn2); hn2_reg[rr] = hn2;
      float hn = clmin(sqrtf(hn2));
      float ath = artanh_c(hn);
      float rgate[4];
#pragma unroll
      for (int c = 0; c < 2; ++c) {
        float wh[4], wn2 = 0.f;
#pragma unroll
        for (int j = 0; j < 4; ++j) { wh[j] = sh_g[r][c * H_ + lane + 32 * j]; wn2 += wh[j] * wh[j]; }
        wn2 = wred(wn2);
        float mxn = clmin(sqrtf(wn2));
        float tv = tanhf(mxn / hn * ath);
        float s = tv / mxn, x2 = tv * tv;
        float xs[4], ux[4], y2 = 0.f, xy = 0.f;
#pragma unroll
        for (int j = 0; j < 4; ++j) {
          xs[j] = wh[j] * s; ux[j] = uxrow[c * H_ + lane + 32 * j];
          y2 += ux[j] * ux[j]; xy += xs[j] * ux[j];
        }
        y2 = wred(y2); xy = wred(xy);
        float ca = 1.f + 2.f * xy + y2, cb = 1.f - x2, den = clmin(1.f + 2.f * xy + x2 * y2);
        float m1[4], x2b = 0.f, xyb = 0.f;
#pragma unroll
        for (int j = 0; j < 4; ++j) {
          m1[j] = (ca * xs[j] + cb * ux[j]) / den;
          float bb = sh_b[c][lane + 32 * j];
          x2b += m1[j] * m1[j]; xyb += m1[j] * bb;
        }
        x2b = wred(x2b); xyb = wred(xyb);
        float y2b = sh_bn2[c];
        float ca2 = 1.f + 2.f * xyb + y2b, cb2 = 1.f - x2b, den2 = clmin(1.f + 2.f * xyb + x2b * y2b);
        float m2[4], n2 = 0.f;
#pragma unroll
        for (int j = 0; j < 4; ++j) {
          m2[j] = (ca2 * m1[j] + cb2 * sh_b[c][lane + 32 * j]) / den2;
          n2 += m2[j] * m2[j];
        }
        n2 = wred(n2);
        float nn = clmin(sqrtf(n2));
        float scg = artanh_c(nn) / nn;
#pragma unroll
        for (int j = 0; j < 4; ++j) {
          float g = 1.f / (1.f + expf(-m2[j] * scg));
          if (c == 0) z_reg[rr][j] = g; else rgate[j] = g;
        }
      }
      // rh = mobius_pointwise_mul(h, r): xn = ||r||
      float rn2 = 0.f, wx[4], wxn2 = 0.f;
#pragma unroll
      for (int j = 0; j < 4; ++j) { rn2 += rgate[j] * rgate[j]; wx[j] = h4[j] * rgate[j]; wxn2 += wx[j] * wx[j]; }
      rn2 = wred(rn2); wxn2 = wred(wxn2);
      float rn = clmin(sqrtf(rn2)), wxn = clmin(sqrtf(wxn2));
      float tvr = tanhf(wxn / rn * artanh_c(rn));
      float srh = tvr / wxn;
      rhn_reg[rr] = clmin(tvr);
#pragma unroll
      for (int j = 0; j < 4; ++j) sh_a[r][lane + 32 * j] = (_Float16)(wx[j] * srh);
    }
    __syncthreads();

    // ---- GEMM2: sh_g[:,0:128] = rh_f16 @ Wh^T (1 tile per wave) ----
    {
      const int nt = wave;
      v8f acc = {};
#pragma unroll
      for (int ks = 0; ks < 4; ++ks) {
        const _Float16* ap = &sh_a[l16][ks * 32 + 16 * g16];
        v8h a0 = *(const v8h*)ap, a1 = *(const v8h*)(ap + 8);
        v16h A;
#pragma unroll
        for (int i = 0; i < 8; ++i) { A[i] = a0[i]; A[8 + i] = a1[i]; }
        acc = __builtin_amdgcn_wmma_f32_16x16x32_f16(false, A, false, Bw2[ks], (short)0, acc, false, false);
      }
#pragma unroll
      for (int i = 0; i < 8; ++i) sh_g[i + 8 * g16][nt * 16 + l16] = acc[i];
    }
    __syncthreads();

    // ---- E2: h_tilde, delta, state update, u/v accumulation ----
#pragma unroll
    for (int rr = 0; rr < 2; ++rr) {
      const int r = r0 + rr, bi = blockIdx.x * 16 + r;
      const float* uxrow = Ux + (size_t)(bi * T_ + t) * N3H + 2 * H_;
      float h4[4];
#pragma unroll
      for (int j = 0; j < 4; ++j) h4[j] = sh_h[r][lane + 32 * j];
      const float hn2 = hn2_reg[rr];
      float whh[4], wn2 = 0.f;
#pragma unroll
      for (int j = 0; j < 4; ++j) { whh[j] = sh_g[r][lane + 32 * j]; wn2 += whh[j] * whh[j]; }
      wn2 = wred(wn2);
      float mxn = clmin(sqrtf(wn2));
      float rhn = rhn_reg[rr];
      float tv = tanhf(mxn / rhn * artanh_c(rhn));
      float s = tv / mxn, x2 = tv * tv;
      float xs[4], ux[4], y2 = 0.f, xy = 0.f;
#pragma unroll
      for (int j = 0; j < 4; ++j) {
        xs[j] = whh[j] * s; ux[j] = uxrow[lane + 32 * j];
        y2 += ux[j] * ux[j]; xy += xs[j] * ux[j];
      }
      y2 = wred(y2); xy = wred(xy);
      float ca = 1.f + 2.f * xy + y2, cb = 1.f - x2, den = clmin(1.f + 2.f * xy + x2 * y2);
      float m1[4], x2b = 0.f, xyb = 0.f;
#pragma unroll
      for (int j = 0; j < 4; ++j) {
        m1[j] = (ca * xs[j] + cb * ux[j]) / den;
        float bb = sh_b[2][lane + 32 * j];
        x2b += m1[j] * m1[j]; xyb += m1[j] * bb;
      }
      x2b = wred(x2b); xyb = wred(xyb);
      float y2b = sh_bn2[2];
      float ca2 = 1.f + 2.f * xyb + y2b, cb2 = 1.f - x2b, den2 = clmin(1.f + 2.f * xyb + x2b * y2b);
      float ht[4], y2t = 0.f, xyd = 0.f;
#pragma unroll
      for (int j = 0; j < 4; ++j) {
        ht[j] = (ca2 * m1[j] + cb2 * sh_b[2][lane + 32 * j]) / den2;
        y2t += ht[j] * ht[j]; xyd += (-h4[j]) * ht[j];
      }
      y2t = wred(y2t); xyd = wred(xyd);
      float cad = 1.f + 2.f * xyd + y2t, cbd = 1.f - hn2, dend = clmin(1.f + 2.f * xyd + hn2 * y2t);
      float dl[4], zn2 = 0.f, wx[4], wxn2 = 0.f;
#pragma unroll
      for (int j = 0; j < 4; ++j) {
        dl[j] = (cad * (-h4[j]) + cbd * ht[j]) / dend;
        float z = z_reg[rr][j];
        zn2 += z * z; wx[j] = dl[j] * z; wxn2 += wx[j] * wx[j];
      }
      zn2 = wred(zn2); wxn2 = wred(wxn2);
      float zn = clmin(sqrtf(zn2)), wxn = clmin(sqrtf(wxn2));
      float tvp = tanhf(wxn / zn * artanh_c(zn));
      float sp = tvp / wxn;
      float pm[4], xyp = 0.f;
#pragma unroll
      for (int j = 0; j < 4; ++j) { pm[j] = wx[j] * sp; xyp += h4[j] * pm[j]; }
      xyp = wred(xyp);
      float y2p = tvp * tvp;
      float can = 1.f + 2.f * xyp + y2p, cbn = 1.f - hn2, denn = clmin(1.f + 2.f * xyp + hn2 * y2p);
      float m1v = mask1[bi * T_ + t], m2v = mask2[bi * T_ + t];
#pragma unroll
      for (int j = 0; j < 4; ++j) {
        float hnw = (can * h4[j] + cbn * pm[j]) / denn;
        u_acc[rr][j] += m1v * hnw;
        v_acc[rr][j] += m2v * hnw;
        sh_h[r][lane + 32 * j] = hnw;
        sh_a[r][lane + 32 * j] = (_Float16)hnw;
      }
    }
    __syncthreads();
  }
#pragma unroll
  for (int rr = 0; rr < 2; ++rr) {
    const int bi = blockIdx.x * 16 + r0 + rr;
#pragma unroll
    for (int j = 0; j < 4; ++j) {
      ubuf[bi * H_ + lane + 32 * j] = u_acc[rr][j];
      vbuf[bi * H_ + lane + 32 * j] = v_acc[rr][j];
    }
  }
}

// ---------------- Kernel 6: head (dist + FF + hyperbolic MLR) ---------------
__device__ __forceinline__ void mobius_add2(const float* x, const float* y, float* o) {
  float x2 = x[0] * x[0] + x[1] * x[1];
  float y2 = y[0] * y[0] + y[1] * y[1];
  float xy = x[0] * y[0] + x[1] * y[1];
  x2 = wred(x2); y2 = wred(y2); xy = wred(xy);
  float ca = 1.f + 2.f * xy + y2, cb = 1.f - x2, den = clmin(1.f + 2.f * xy + x2 * y2);
  float o0 = (ca * x[0] + cb * y[0]) / den, o1 = (ca * x[1] + cb * y[1]) / den;
  o[0] = o0; o[1] = o1;
}

__global__ __launch_bounds__(256) void k_head(const float* __restrict__ ubuf, const float* __restrict__ vbuf,
                                              const float* __restrict__ Wu, const float* __restrict__ Wv,
                                              const float* __restrict__ bff, const float* __restrict__ bffd,
                                              const float* __restrict__ Wc, const float* __restrict__ cs_emb,
                                              const int* __restrict__ cids,
                                              const float* __restrict__ pmlr, const float* __restrict__ amlr,
                                              float* __restrict__ out) {
  __shared__ float sh_u[8][H_];
  __shared__ float sh_v[8][H_];
  __shared__ float sh_c[8][CSD];
  const int wave = threadIdx.x >> 5, lane = threadIdx.x & 31;
  const int bi = blockIdx.x * 8 + wave;
  const int o0 = lane, o1 = lane + 32;

  float uu[4], vv[4]; float un2 = 0.f, vn2 = 0.f, uvd = 0.f;
#pragma unroll
  for (int j = 0; j < 4; ++j) {
    uu[j] = ubuf[bi * H_ + lane + 32 * j]; vv[j] = vbuf[bi * H_ + lane + 32 * j];
    sh_u[wave][lane + 32 * j] = uu[j]; sh_v[wave][lane + 32 * j] = vv[j];
    un2 += uu[j] * uu[j]; vn2 += vv[j] * vv[j]; uvd += uu[j] * vv[j];
  }
  un2 = wred(un2); vn2 = wred(vn2); uvd = wred(uvd);

  // dist(u,v) = 2*artanh(||mobius_add(-u,v)||)
  float xy = -uvd;
  float ca = 1.f + 2.f * xy + vn2, cb = 1.f - un2, den = clmin(1.f + 2.f * xy + un2 * vn2);
  float dn2 = 0.f;
#pragma unroll
  for (int j = 0; j < 4; ++j) { float d = (ca * (-uu[j]) + cb * vv[j]) / den; dn2 += d * d; }
  dn2 = wred(dn2);
  float dist = 2.f * artanh_c(sqrtf(dn2));

  // mobius_matvec(W_ff_u, u) and (W_ff_v, v): 2 output dims per lane
  float mu[2] = {0.f, 0.f}, mv[2] = {0.f, 0.f};
  for (int k = 0; k < H_; ++k) {
    float uk = sh_u[wave][k], vk = sh_v[wave][k];
    mu[0] += Wu[o0 * H_ + k] * uk; mu[1] += Wu[o1 * H_ + k] * uk;
    mv[0] += Wv[o0 * H_ + k] * vk; mv[1] += Wv[o1 * H_ + k] * vk;
  }
  float mn2 = wred(mu[0] * mu[0] + mu[1] * mu[1]);
  float xnu = clmin(sqrtf(un2)), mxn = clmin(sqrtf(mn2));
  float su = tanhf(mxn / xnu * artanh_c(xnu)) / mxn;
  mu[0] *= su; mu[1] *= su;
  mn2 = wred(mv[0] * mv[0] + mv[1] * mv[1]);
  float xnv = clmin(sqrtf(vn2)); mxn = clmin(sqrtf(mn2));
  float sv = tanhf(mxn / xnv * artanh_c(xnv)) / mxn;
  mv[0] *= sv; mv[1] *= sv;

  float acc2[2];
  mobius_add2(mu, mv, acc2);
  float b2[2] = { bff[o0], bff[o1] };
  mobius_add2(acc2, b2, acc2);

  // mobius_scalar_mul(dist, b_ff_d)
  float bd[2] = { bffd[o0], bffd[o1] };
  float bdn2 = wred(bd[0] * bd[0] + bd[1] * bd[1]);
  float xnd = clmin(sqrtf(bdn2));
  float tvd = tanhf(dist * artanh_c(xnd));
  float sm[2] = { bd[0] * tvd / xnd, bd[1] * tvd / xnd };
  mobius_add2(acc2, sm, acc2);

  // common-sense embedding path
  int cid = cids[bi];
  float cv[2] = { cs_emb[cid * CSD + o0], cs_emb[cid * CSD + o1] };
  sh_c[wave][o0] = cv[0]; sh_c[wave][o1] = cv[1];
  float cn2 = wred(cv[0] * cv[0] + cv[1] * cv[1]);
  float mc[2] = {0.f, 0.f};
  for (int k = 0; k < CSD; ++k) {
    float ck = sh_c[wave][k];
    mc[0] += Wc[o0 * CSD + k] * ck; mc[1] += Wc[o1 * CSD + k] * ck;
  }
  float mcn2 = wred(mc[0] * mc[0] + mc[1] * mc[1]);
  float xnc = clmin(sqrtf(cn2)), mxnc = clmin(sqrtf(mcn2));
  float scc = tanhf(mxnc / xnc * artanh_c(xnc)) / mxnc;
  mc[0] *= scc; mc[1] *= scc;
  mobius_add2(acc2, mc, acc2);

  // logmap0 then expmap0
  float an2 = wred(acc2[0] * acc2[0] + acc2[1] * acc2[1]);
  float an = clmin(sqrtf(an2));
  float sl = artanh_c(an) / an;
  acc2[0] *= sl; acc2[1] *= sl;
  float ln2 = wred(acc2[0] * acc2[0] + acc2[1] * acc2[1]);
  float ln = clmin(sqrtf(ln2));
  float se = tanhf(ln) / ln;
  acc2[0] *= se; acc2[1] *= se;

  // hyperbolic MLR over 4 classes
  for (int cls = 0; cls < NCLS; ++cls) {
    float p2[2] = { -pmlr[cls * DOUT_ + o0], -pmlr[cls * DOUT_ + o1] };
    float mpx[2];
    mobius_add2(p2, acc2, mpx);
    float mp2 = wred(mpx[0] * mpx[0] + mpx[1] * mpx[1]);
    float lam = 2.f / (1.f - mp2);
    float a0 = amlr[cls * DOUT_ + o0], a1 = amlr[cls * DOUT_ + o1];
    float na2 = wred(a0 * a0 + a1 * a1);
    float na = sqrtf(na2);
    float nac = fmaxf(na, 1e-12f);
    float pxa = wred(mpx[0] * a0 / nac + mpx[1] * a1 / nac);
    if (lane == 0) out[bi * NCLS + cls] = 2.f * na * asinhf(pxa * lam);
  }
}

extern "C" void kernel_launch(void* const* d_in, const int* in_sizes, int n_in,
                              void* d_out, int out_size, void* d_ws, size_t ws_size,
                              hipStream_t stream) {
  (void)in_sizes; (void)n_in; (void)out_size; (void)ws_size;
  const float* seq   = (const float*)d_in[0];
  const float* mask1 = (const float*)d_in[1];
  const float* mask2 = (const float*)d_in[2];
  const int*   cids  = (const int*)d_in[3];
  const float* csem  = (const float*)d_in[4];
  const float* wz = (const float*)d_in[5];
  const float* wr = (const float*)d_in[6];
  const float* wh = (const float*)d_in[7];
  const float* uz = (const float*)d_in[8];
  const float* ur = (const float*)d_in[9];
  const float* uh = (const float*)d_in[10];
  const float* bz = (const float*)d_in[11];
  const float* br = (const float*)d_in[12];
  const float* bh = (const float*)d_in[13];
  const float* Wfu = (const float*)d_in[14];
  const float* Wfv = (const float*)d_in[15];
  const float* bff = (const float*)d_in[16];
  const float* bffd = (const float*)d_in[17];
  const float* Wfc = (const float*)d_in[18];
  const float* pmlr = (const float*)d_in[19];
  const float* amlr = (const float*)d_in[20];

  char* ws = (char*)d_ws;
  size_t off = 0;
  _Float16* projh = (_Float16*)(ws + off); off += (size_t)B_ * T_ * DIN * 2;   // 50.3 MB
  float*    xn    = (float*)(ws + off);    off += (size_t)B_ * T_ * 4;          // 128 KB
  _Float16* Ucat  = (_Float16*)(ws + off); off += (size_t)N3H * DIN * 2;        // 576 KB
  _Float16* Wcat  = (_Float16*)(ws + off); off += (size_t)N3H * H_ * 2;         // 96 KB
  float*    Ux    = (float*)(ws + off);    off += (size_t)B_ * T_ * N3H * 4;    // 50.3 MB
  float*    ubuf  = (float*)(ws + off);    off += (size_t)B_ * H_ * 4;
  float*    vbuf  = (float*)(ws + off);    off += (size_t)B_ * H_ * 4;

  k_prep<<<4096, 256, 0, stream>>>(seq, projh, xn);
  k_pack<<<1344, 256, 0, stream>>>(wz, wr, wh, uz, ur, uh, Ucat, Wcat);
  k_gemm_ux<<<2048, 256, 0, stream>>>(projh, Ucat, Ux);
  k_scale_ux<<<12288, 256, 0, stream>>>(Ux, xn);
  k_scan<<<16, 256, 0, stream>>>(Ux, Wcat, bz, br, bh, mask1, mask2, ubuf, vbuf);
  k_head<<<32, 256, 0, stream>>>(ubuf, vbuf, Wfu, Wfv, bff, bffd, Wfc, csem, cids, pmlr, amlr,
                                 (float*)d_out);
}
